// SelfAttention_90314572300799
// MI455X (gfx1250) — compile-verified, matrix operand order checked by reading, not run on hardware
//
#include <hip/hip_runtime.h>

// ---------------------------------------------------------------------------
// Causal self-attention for MI455X (gfx1250, wave32).
//   B=4, S=2048, D=1024. All GEMMs: v_wmma_f32_16x16x32_bf16, double-buffered
//   LDS pipeline fed by GLOBAL_LOAD_ASYNC_TO_LDS_B128 (ASYNCcnt) when available.
// ---------------------------------------------------------------------------

typedef __bf16 bf16_t;
typedef bf16_t v16bf __attribute__((ext_vector_type(16)));
typedef bf16_t v8bf  __attribute__((ext_vector_type(8)));
typedef bf16_t v4bf  __attribute__((ext_vector_type(4)));
typedef float  v8f   __attribute__((ext_vector_type(8)));
typedef int    v4i   __attribute__((ext_vector_type(4)));

#define D_MODEL 1024
#define SEQ     2048
#define NBATCH  4
#define MROWS   (NBATCH * SEQ)      // 8192 flattened (batch, seq) rows

#define TILE_ELEMS (128 * 40)       // one padded 128x32 bf16 LDS tile

#if defined(__has_builtin)
# if __has_builtin(__builtin_amdgcn_global_load_async_to_lds_b128)
#  define USE_ASYNC_LDS 1
# endif
#endif
#ifndef USE_ASYNC_LDS
# define USE_ASYNC_LDS 0
#endif

#define AS1 __attribute__((address_space(1)))
#define AS3 __attribute__((address_space(3)))

#if USE_ASYNC_LDS
# if __has_builtin(__builtin_amdgcn_s_wait_asynccnt)
#  define WAIT_ASYNC(n) __builtin_amdgcn_s_wait_asynccnt(n)
# else
#  define WAIT_ASYNC(n) asm volatile("s_wait_asynccnt %0" :: "i"(n) : "memory")
# endif
#else
# define WAIT_ASYNC(n) ((void)0)
#endif

// float -> bf16, round-to-nearest-even (bit math)
__device__ __forceinline__ bf16_t f2bf(float f) {
  unsigned u = __builtin_bit_cast(unsigned, f);
  unsigned r = u + 0x7FFFu + ((u >> 16) & 1u);
  unsigned short h = (unsigned short)(r >> 16);
  return __builtin_bit_cast(bf16_t, h);
}

#define WMMA_BF16(A_, B_, C_) \
  __builtin_amdgcn_wmma_f32_16x16x32_bf16(false, (A_), false, (B_), (short)0, (C_), false, false)

// Copy 16 contiguous bf16 (32B) global -> LDS for this lane.
// Async path: two GLOBAL_LOAD_ASYNC_TO_LDS_B128 (2 ASYNCcnt increments/wave).
__device__ __forceinline__ void copy16_g2l(bf16_t* ldst, const bf16_t* gsrc) {
#if USE_ASYNC_LDS
  __builtin_amdgcn_global_load_async_to_lds_b128(
      (AS1 v4i*)(unsigned long long)gsrc,
      (AS3 v4i*)(unsigned)(unsigned long long)ldst, 0, 0);
  __builtin_amdgcn_global_load_async_to_lds_b128(
      (AS1 v4i*)(unsigned long long)(gsrc + 8),
      (AS3 v4i*)(unsigned)(unsigned long long)(ldst + 8), 0, 0);
#else
  const v8bf* g = (const v8bf*)gsrc;
  *(v8bf*)ldst       = g[0];
  *(v8bf*)(ldst + 8) = g[1];
#endif
}

// Per-lane A/B fragment for 16x16x32 bf16 (wave32):
// lane = half*16 + lm;  e<8 -> k = half*8+e ; e>=8 -> k = 16+half*8+(e-8)
// i.e. two contiguous 16B chunks of one row.
__device__ __forceinline__ v16bf load_frag_row(const bf16_t* rowbase, int half) {
  v8bf lo = *(const v8bf*)(rowbase + half * 8);
  v8bf hi = *(const v8bf*)(rowbase + 16 + half * 8);
  v16bf r;
#pragma unroll
  for (int i = 0; i < 8; ++i) { r[i] = lo[i]; r[i + 8] = hi[i]; }
  return r;
}

// Stage a 128x32 bf16 tile (row-major global, leading dim ldg) into LDS (stride 40).
__device__ __forceinline__ void stage_tileA(bf16_t* lds, const bf16_t* gsrc,
                                            int ldg, int k0, int tid) {
  int row = tid >> 1;
  int seg = (tid & 1) * 16;
  copy16_g2l(lds + row * 40 + seg, gsrc + (size_t)row * ldg + k0 + seg);
}

// Stage a 32x128 V tile TRANSPOSED into LDS as [d][k] (stride 40), so B
// fragments become vectorized row loads. lane = key row => ~2-way LDS bank
// conflict on the 2B scatter stores, amortized once per k-step.
__device__ __forceinline__ void stage_tileVt(bf16_t* lds, const bf16_t* Vp,
                                             int k0, int n0, int tid) {
  int krow = tid & 31;            // key within tile
  int seg  = (tid >> 5) * 16;     // d segment
  const v8bf* g = (const v8bf*)(Vp + (size_t)(k0 + krow) * D_MODEL + n0 + seg);
  v8bf x0 = g[0], x1 = g[1];
#pragma unroll
  for (int j = 0; j < 8; ++j) {
    lds[(seg + j)     * 40 + krow] = x0[j];
    lds[(seg + 8 + j) * 40 + krow] = x1[j];
  }
}

// Double-buffered 128x128 GEMM pipeline: C += A(128 x 32k) * B(32k x 128),
// both operands row-major-by-reduction (row r holds k-run). 8 waves of 32x64.
__device__ __forceinline__ void gemm_tile_pipeline(
    const bf16_t* __restrict__ Arow, const bf16_t* __restrict__ Brow,
    int ldg, int nsteps, bf16_t* As, bf16_t* Bs, int tid, v8f (&c)[2][4]) {
  const int lane = tid & 31;
  const int lm = lane & 15, half = lane >> 4;
  const int wid = tid >> 5;
  const int wm = wid & 3, wn = wid >> 2;

  stage_tileA(As, Arow, ldg, 0, tid);
  stage_tileA(Bs, Brow, ldg, 0, tid);
  for (int i = 0; i < nsteps; ++i) {
    const bf16_t* Ac = As + (i & 1) * TILE_ELEMS;
    const bf16_t* Bc = Bs + (i & 1) * TILE_ELEMS;
    if (i + 1 < nsteps) {
      int nxt = ((i + 1) & 1) * TILE_ELEMS;
      stage_tileA(As + nxt, Arow, ldg, (i + 1) << 5, tid);
      stage_tileA(Bs + nxt, Brow, ldg, (i + 1) << 5, tid);
      if (i + 2 < nsteps)   // L2 prefetch two tiles ahead (global_prefetch_b8)
        __builtin_prefetch(Arow + (size_t)(tid >> 1) * ldg + ((i + 2) << 5), 0, 1);
      WAIT_ASYNC(4);        // drain tile i's 4 async ops; keep tile i+1 in flight
    } else {
      WAIT_ASYNC(0);
    }
    __syncthreads();
    v16bf a[2], bb[4];
#pragma unroll
    for (int mt = 0; mt < 2; ++mt)
      a[mt] = load_frag_row(Ac + (wm * 32 + mt * 16 + lm) * 40, half);
#pragma unroll
    for (int nt = 0; nt < 4; ++nt)
      bb[nt] = load_frag_row(Bc + (wn * 64 + nt * 16 + lm) * 40, half);
#pragma unroll
    for (int mt = 0; mt < 2; ++mt)
#pragma unroll
      for (int nt = 0; nt < 4; ++nt)
        c[mt][nt] = WMMA_BF16(a[mt], bb[nt], c[mt][nt]);
    __syncthreads();
  }
}

// ---------------------------------------------------------------------------
// fp32 -> bf16 bulk convert (vectorized x4)
// ---------------------------------------------------------------------------
__global__ __launch_bounds__(256) void cvt_f32_bf16_x4(const float4* __restrict__ s,
                                                       v4bf* __restrict__ d, int n4) {
  int i = blockIdx.x * blockDim.x + threadIdx.x;
  int stride = gridDim.x * blockDim.x;
  for (; i < n4; i += stride) {
    float4 v = s[i];
    v4bf o;
    o[0] = f2bf(v.x); o[1] = f2bf(v.y); o[2] = f2bf(v.z); o[3] = f2bf(v.w);
    d[i] = o;
  }
}

// ---------------------------------------------------------------------------
// QKV projection: Out[m,n] = sum_k X[m,k]*W[n,k] + bias[n]  (x @ W.T + b)
// ---------------------------------------------------------------------------
__global__ __launch_bounds__(256) void qkv_gemm_kernel(
    const bf16_t* __restrict__ Xb,
    const bf16_t* __restrict__ Wqb, const bf16_t* __restrict__ Wkb,
    const bf16_t* __restrict__ Wvb,
    const float* __restrict__ Bq, const float* __restrict__ Bk,
    const float* __restrict__ Bv,
    bf16_t* __restrict__ Qb, bf16_t* __restrict__ Kb, bf16_t* __restrict__ Vb) {
  __shared__ alignas(16) bf16_t As[2 * TILE_ELEMS];
  __shared__ alignas(16) bf16_t Bs[2 * TILE_ELEMS];

  const bf16_t* W; const float* bias; bf16_t* Out;
  if (blockIdx.z == 0)      { W = Wqb; bias = Bq; Out = Qb; }
  else if (blockIdx.z == 1) { W = Wkb; bias = Bk; Out = Kb; }
  else                      { W = Wvb; bias = Bv; Out = Vb; }

  const int tid = threadIdx.x;
  const int wid = tid >> 5, lane = tid & 31;
  const int lm = lane & 15, half = lane >> 4;
  const int wm = wid & 3, wn = wid >> 2;
  const int m0 = blockIdx.x * 128, n0 = blockIdx.y * 128;

  v8f c[2][4] = {};
  gemm_tile_pipeline(Xb + (size_t)m0 * D_MODEL, W + (size_t)n0 * D_MODEL,
                     D_MODEL, D_MODEL / 32, As, Bs, tid, c);
#pragma unroll
  for (int mt = 0; mt < 2; ++mt)
#pragma unroll
    for (int nt = 0; nt < 4; ++nt) {
      int n = n0 + wn * 64 + nt * 16 + lm;
      float bsv = bias[n];
#pragma unroll
      for (int r = 0; r < 8; ++r) {
        int m = m0 + wm * 32 + mt * 16 + half * 8 + r;
        Out[(size_t)m * D_MODEL + n] = f2bf(c[mt][nt][r] + bsv);
      }
    }
}

// ---------------------------------------------------------------------------
// Scores: S[q,k] = (Q[q,:] . K[k,:]) / 32, causal (-3e38 above diagonal).
// Blocks fully above the diagonal skip the GEMM entirely.
// ---------------------------------------------------------------------------
__global__ __launch_bounds__(256) void scores_kernel(
    const bf16_t* __restrict__ Qb, const bf16_t* __restrict__ Kb,
    float* __restrict__ S) {
  __shared__ alignas(16) bf16_t As[2 * TILE_ELEMS];
  __shared__ alignas(16) bf16_t Bs[2 * TILE_ELEMS];

  const int bz = blockIdx.z;
  const bf16_t* Q = Qb + (size_t)bz * SEQ * D_MODEL;
  const bf16_t* K = Kb + (size_t)bz * SEQ * D_MODEL;
  float* Sb = S + (size_t)bz * SEQ * SEQ;

  const int m0 = blockIdx.x * 128, n0 = blockIdx.y * 128;
  const int tid = threadIdx.x;

  if (n0 > m0 + 127) {  // entirely masked block: just fill
    for (int i = tid; i < 128 * 128; i += 256) {
      int r = i >> 7, cc = i & 127;
      Sb[(size_t)(m0 + r) * SEQ + (n0 + cc)] = -3.0e38f;
    }
    return;
  }

  const int wid = tid >> 5, lane = tid & 31;
  const int lm = lane & 15, half = lane >> 4;
  const int wm = wid & 3, wn = wid >> 2;

  v8f c[2][4] = {};
  gemm_tile_pipeline(Q + (size_t)m0 * D_MODEL, K + (size_t)n0 * D_MODEL,
                     D_MODEL, D_MODEL / 32, As, Bs, tid, c);

  const float scale = 0.03125f;  // 1/sqrt(1024)
#pragma unroll
  for (int mt = 0; mt < 2; ++mt)
#pragma unroll
    for (int nt = 0; nt < 4; ++nt) {
      int n = n0 + wn * 64 + nt * 16 + lm;
#pragma unroll
      for (int r = 0; r < 8; ++r) {
        int m = m0 + wm * 32 + mt * 16 + half * 8 + r;
        float v = c[mt][nt][r] * scale;
        if (n > m) v = -3.0e38f;
        Sb[(size_t)m * SEQ + n] = v;
      }
    }
}

// ---------------------------------------------------------------------------
// Row softmax over 2048 entries (masked entries already -3e38), bf16 output.
// ---------------------------------------------------------------------------
__global__ __launch_bounds__(256) void softmax_kernel(const float* __restrict__ S,
                                                      bf16_t* __restrict__ P) {
  __shared__ float red[256];
  const int row = blockIdx.x, bz = blockIdx.y, tid = threadIdx.x;
  const float* s = S + ((size_t)bz * SEQ + row) * SEQ;
  bf16_t* p = P + ((size_t)bz * SEQ + row) * SEQ;

  float v[8];
  float mx = -3.4e38f;
#pragma unroll
  for (int i = 0; i < 8; ++i) { v[i] = s[tid + i * 256]; mx = fmaxf(mx, v[i]); }
  red[tid] = mx; __syncthreads();
  for (int off = 128; off > 0; off >>= 1) {
    if (tid < off) red[tid] = fmaxf(red[tid], red[tid + off]);
    __syncthreads();
  }
  mx = red[0]; __syncthreads();

  float sum = 0.f;
#pragma unroll
  for (int i = 0; i < 8; ++i) { v[i] = __expf(v[i] - mx); sum += v[i]; }
  red[tid] = sum; __syncthreads();
  for (int off = 128; off > 0; off >>= 1) {
    if (tid < off) red[tid] += red[tid + off];
    __syncthreads();
  }
  float inv = 1.f / red[0];
#pragma unroll
  for (int i = 0; i < 8; ++i) p[tid + i * 256] = f2bf(v[i] * inv);
}

// ---------------------------------------------------------------------------
// O = P @ V  (keys = reduction dim). Causal structure truncates the key loop
// at m0+128. P staged async; V staged transposed (sync scatter stores).
// ---------------------------------------------------------------------------
__global__ __launch_bounds__(256) void pv_kernel(
    const bf16_t* __restrict__ P, const bf16_t* __restrict__ V,
    float* __restrict__ O) {
  __shared__ alignas(16) bf16_t Ps[2 * TILE_ELEMS];
  __shared__ alignas(16) bf16_t Vt[2 * TILE_ELEMS];

  const int bz = blockIdx.z;
  const bf16_t* Pp = P + (size_t)bz * SEQ * SEQ;
  const bf16_t* Vp = V + (size_t)bz * SEQ * D_MODEL;
  float* Op = O + (size_t)bz * SEQ * D_MODEL;

  const int m0 = blockIdx.x * 128, n0 = blockIdx.y * 128;
  const int tid = threadIdx.x;
  const int wid = tid >> 5, lane = tid & 31;
  const int lm = lane & 15, half = lane >> 4;
  const int wm = wid & 3, wn = wid >> 2;

  const bf16_t* Prow = Pp + (size_t)m0 * SEQ;
  const int nsteps = (m0 + 128) >> 5;   // keys needed: [0, m0+128)

  v8f c[2][4] = {};
  stage_tileA(Ps, Prow, SEQ, 0, tid);
  stage_tileVt(Vt, Vp, 0, n0, tid);
  for (int i = 0; i < nsteps; ++i) {
    const bf16_t* Pc = Ps + (i & 1) * TILE_ELEMS;
    const bf16_t* Vc = Vt + (i & 1) * TILE_ELEMS;
    if (i + 1 < nsteps) {
      int nxt = ((i + 1) & 1) * TILE_ELEMS;
      stage_tileA(Ps + nxt, Prow, SEQ, (i + 1) << 5, tid);
      stage_tileVt(Vt + nxt, Vp, (i + 1) << 5, n0, tid);
      if (i + 2 < nsteps)
        __builtin_prefetch(Vp + (size_t)(((i + 2) << 5) + (tid & 31)) * D_MODEL + n0, 0, 1);
      WAIT_ASYNC(2);      // drain tile i's 2 async P ops
    } else {
      WAIT_ASYNC(0);
    }
    __syncthreads();
    v16bf a[2], bb[4];
#pragma unroll
    for (int mt = 0; mt < 2; ++mt)
      a[mt] = load_frag_row(Pc + (wm * 32 + mt * 16 + lm) * 40, half);
#pragma unroll
    for (int nt = 0; nt < 4; ++nt)
      bb[nt] = load_frag_row(Vc + (wn * 64 + nt * 16 + lm) * 40, half);
#pragma unroll
    for (int mt = 0; mt < 2; ++mt)
#pragma unroll
      for (int nt = 0; nt < 4; ++nt)
        c[mt][nt] = WMMA_BF16(a[mt], bb[nt], c[mt][nt]);
    __syncthreads();
  }
#pragma unroll
  for (int mt = 0; mt < 2; ++mt)
#pragma unroll
    for (int nt = 0; nt < 4; ++nt) {
      int n = n0 + wn * 64 + nt * 16 + lm;
#pragma unroll
      for (int r = 0; r < 8; ++r) {
        int m = m0 + wm * 32 + mt * 16 + half * 8 + r;
        Op[(size_t)m * D_MODEL + n] = c[mt][nt][r];
      }
    }
}

// ---------------------------------------------------------------------------
extern "C" void kernel_launch(void* const* d_in, const int* in_sizes, int n_in,
                              void* d_out, int out_size, void* d_ws, size_t ws_size,
                              hipStream_t stream) {
  (void)in_sizes; (void)n_in; (void)out_size; (void)ws_size;
  const float* X  = (const float*)d_in[0];
  const float* Wq = (const float*)d_in[1];
  const float* bq = (const float*)d_in[2];
  const float* Wk = (const float*)d_in[3];
  const float* bk = (const float*)d_in[4];
  const float* Wv = (const float*)d_in[5];
  const float* bv = (const float*)d_in[6];
  float* out = (float*)d_out;

  char* ws = (char*)d_ws;
  const size_t MB = 1024u * 1024u;
  bf16_t* Xb  = (bf16_t*)(ws + 0);        // 16 MB  (8192 x 1024 bf16)
  bf16_t* Wqb = (bf16_t*)(ws + 16 * MB);  //  2 MB
  bf16_t* Wkb = (bf16_t*)(ws + 18 * MB);  //  2 MB
  bf16_t* Wvb = (bf16_t*)(ws + 20 * MB);  //  2 MB
  bf16_t* Qb  = (bf16_t*)(ws + 22 * MB);  // 16 MB
  bf16_t* Kb  = (bf16_t*)(ws + 38 * MB);  // 16 MB
  bf16_t* Vb  = (bf16_t*)(ws + 54 * MB);  // 16 MB
  float*  S   = (float*) (ws + 70 * MB);  // 64 MB  (4 x 2048 x 2048 f32)
  bf16_t* Pr  = (bf16_t*)(ws + 134 * MB); // 32 MB  (bf16 probabilities)

  // 1) bf16 conversions
  cvt_f32_bf16_x4<<<2048, 256, 0, stream>>>((const float4*)X, (v4bf*)Xb,
                                            (MROWS * D_MODEL) / 4);
  cvt_f32_bf16_x4<<<512, 256, 0, stream>>>((const float4*)Wq, (v4bf*)Wqb,
                                           (D_MODEL * D_MODEL) / 4);
  cvt_f32_bf16_x4<<<512, 256, 0, stream>>>((const float4*)Wk, (v4bf*)Wkb,
                                           (D_MODEL * D_MODEL) / 4);
  cvt_f32_bf16_x4<<<512, 256, 0, stream>>>((const float4*)Wv, (v4bf*)Wvb,
                                           (D_MODEL * D_MODEL) / 4);

  // 2) Q/K/V projections (fused over blockIdx.z)
  qkv_gemm_kernel<<<dim3(MROWS / 128, D_MODEL / 128, 3), 256, 0, stream>>>(
      Xb, Wqb, Wkb, Wvb, bq, bk, bv, Qb, Kb, Vb);

  // 3) causal scores
  scores_kernel<<<dim3(SEQ / 128, SEQ / 128, NBATCH), 256, 0, stream>>>(Qb, Kb, S);

  // 4) softmax -> bf16 probabilities
  softmax_kernel<<<dim3(SEQ, NBATCH), 256, 0, stream>>>(S, Pr);

  // 5) O = P @ V
  pv_kernel<<<dim3(SEQ / 128, D_MODEL / 128, NBATCH), 256, 0, stream>>>(Pr, Vb, out);
}
